// CondHandwritingGenerator_30107720745090
// MI455X (gfx1250) — matrix operand order repaired and checked by reading, not compiled, not run on hardware
//
#include <hip/hip_runtime.h>
#include <hip/hip_bf16.h>
#include <math.h>

// ---------------------------------------------------------------------------
// CDNA5 / gfx1250 implementation of the conditional handwriting generator.
// WMMA (bf16 -> f32) for all parallel GEMMs with 16x64 register blocking;
// persistent 8-WGP LDS-resident recurrence for the three LSTM scans;
// deterministic tree reductions for the mixture-density loss.
// ---------------------------------------------------------------------------

#define T_STEPS 4096
#define HID     512
#define GATES   2048
#define NC      80
#define U_LEN   512
#define KMIX    10
#define MMIX    20
#define EPS_F   1e-8f

typedef __bf16 v16bf __attribute__((ext_vector_type(16)));
typedef __bf16 v8bf  __attribute__((ext_vector_type(8)));
typedef float  v8f   __attribute__((ext_vector_type(8)));

__device__ __forceinline__ unsigned short f2bf(float f) {
    unsigned int u = __float_as_uint(f);
    u += 0x7FFFu + ((u >> 16) & 1u);            // round-to-nearest-even
    return (unsigned short)(u >> 16);
}
__device__ __forceinline__ float sigmoidf(float x) { return 1.f / (1.f + __expf(-x)); }

// ---------------------------------------------------------------------------
// WMMA GEMM:  D[M x N] = A_bf16[M x K] * B_bf16[N x K]^T (+ bias[n]) (+ Cadd)
// One wave computes a 16 x (16*NT) strip: the A fragment is loaded once per
// k-step and reused for NT WMMAs (NT-way register blocking). 4 waves / block.
// All dims multiples of 16, K multiple of 32. Emits v_wmma_f32_16x16x32_bf16.
// ---------------------------------------------------------------------------
template <int NT>
__global__ void wmma_gemm(const unsigned short* __restrict__ A, int lda,
                          const unsigned short* __restrict__ B, int ldb,
                          const float* __restrict__ bias,
                          const float* __restrict__ Cadd, int ldc,
                          float* __restrict__ D, int ldd,
                          int Ngroups, int K)          // Ngroups = Ntiles / NT
{
    const int lane = threadIdx.x & 31;
    const int wave = threadIdx.x >> 5;
    const int wt   = blockIdx.x * 4 + wave;
    const int tm   = wt / Ngroups;
    const int tg   = wt - tm * Ngroups;
    const int half = lane >> 4;      // 0: K 0..7 / 16..23   1: K 8..15 / 24..31
    const int l15  = lane & 15;

    // A fragment: lane holds row (tm*16+l15), two contiguous 8-half groups.
    const unsigned short* a0 = A + (size_t)(tm * 16 + l15) * lda + half * 8;
    // B fragments: lane holds column n (row of B storage), two contiguous 8-half groups.
    const unsigned short* b0 = B + (size_t)(tg * NT * 16 + l15) * ldb + half * 8;

    v8f c[NT];
    v8f zero = {};
#pragma unroll
    for (int j = 0; j < NT; ++j) c[j] = zero;

    for (int k = 0; k < K; k += 32) {
        union { v16bf v; v8bf h[2]; } af;
        af.h[0] = *(const v8bf*)(a0 + k);
        af.h[1] = *(const v8bf*)(a0 + k + 16);
#pragma unroll
        for (int j = 0; j < NT; ++j) {
            union { v16bf v; v8bf h[2]; } bu;
            bu.h[0] = *(const v8bf*)(b0 + (size_t)j * 16 * ldb + k);
            bu.h[1] = *(const v8bf*)(b0 + (size_t)j * 16 * ldb + k + 16);
            v16bf bfv = bu.v;
            c[j] = __builtin_amdgcn_wmma_f32_16x16x32_bf16(false, af.v, false, bfv,
                                                           (short)0, c[j], false, false);
        }
    }
    // C/D layout: VGPR v -> M = v + 8*half, N = l15.
    const int mbase = tm * 16 + half * 8;
#pragma unroll
    for (int j = 0; j < NT; ++j) {
        const int n = (tg * NT + j) * 16 + l15;
        float bv = bias ? bias[n] : 0.f;
#pragma unroll
        for (int v = 0; v < 8; ++v) {
            float d = c[j][v] + bv;
            if (Cadd) d += Cadd[(size_t)(mbase + v) * ldc + n];
            D[(size_t)(mbase + v) * ldd + n] = d;
        }
    }
}

// ---------------------------------------------------------------------------
// Persistent LSTM recurrence. 8 workgroups, each owns 64 hidden units and
// keeps its 256 x 512 slice of whh^T in LDS as packed bf16 pairs (256 KB,
// transposed [k][gate] layout -> conflict-free). Per step: matvec from LDS,
// gate nonlinearity, h broadcast via L2 + atomic release/acquire sync.
// ---------------------------------------------------------------------------
__global__ void lstm_scan(const float* __restrict__ xg,     // T x 2048 (both biases folded)
                          const float* __restrict__ whh,    // 2048 x 512 (f32)
                          float* __restrict__ h_all,        // T x 512
                          volatile float* __restrict__ h_cur,   // 512 broadcast buffer
                          volatile unsigned int* __restrict__ sync)
{
    extern __shared__ char smem[];
    unsigned int* lds_w = (unsigned int*)smem;               // [256 kp][256 gate] packed 2xbf16
    float* lds_h = (float*)(smem + 256 * 256 * 4);           // 512
    float* lds_g = lds_h + 512;                              // 256

    const int tid  = threadIdx.x;        // 0..255
    const int wg   = blockIdx.x;         // 0..7
    const int type = tid >> 6;           // 0=i 1=f 2=g 3=o
    const int lu   = tid & 63;
    const int unit = wg * 64 + lu;       // hidden unit owned (for type==0 threads)
    const int grow = type * 512 + unit;  // gate row in 0..2047

    const float* wr = whh + (size_t)grow * 512;
    for (int kp = 0; kp < 256; ++kp) {
        unsigned int lo = f2bf(wr[2 * kp]);
        unsigned int hi = f2bf(wr[2 * kp + 1]);
        lds_w[kp * 256 + tid] = (hi << 16) | lo;
    }
    lds_h[tid] = 0.f;
    lds_h[256 + tid] = 0.f;
    __syncthreads();

    float cstate = 0.f;                  // valid for tid < 64
    for (int t = 0; t < T_STEPS; ++t) {
        float acc = xg[(size_t)t * GATES + grow];
        if (t + 1 < T_STEPS)             // prefetch next step's gate preactivation
            __builtin_prefetch(xg + (size_t)(t + 1) * GATES + grow, 0, 3);
#pragma unroll 4
        for (int kp = 0; kp < 256; ++kp) {
            unsigned int pw = lds_w[kp * 256 + tid];
            float h0 = lds_h[2 * kp];
            float h1 = lds_h[2 * kp + 1];
            acc += h0 * __uint_as_float(pw << 16);
            acc += h1 * __uint_as_float(pw & 0xFFFF0000u);
        }
        lds_g[tid] = acc;
        __syncthreads();
        if (tid < 64) {
            float gi = lds_g[lu];
            float gf = lds_g[64 + lu];
            float gg = lds_g[128 + lu];
            float go = lds_g[192 + lu];
            cstate = sigmoidf(gf) * cstate + sigmoidf(gi) * tanhf(gg);
            float h = sigmoidf(go) * tanhf(cstate);
            h_all[(size_t)t * HID + unit] = h;
            h_cur[unit] = h;
        }
        __threadfence();                 // release h_cur
        __syncthreads();
        if (tid == 0) atomicAdd((unsigned int*)sync, 1u);
        if (t + 1 < T_STEPS) {
            if (tid == 0) {
                while (*sync < 8u * (unsigned)(t + 1)) { __builtin_amdgcn_s_sleep(1); }
            }
            __syncthreads();
            __threadfence();             // acquire
            lds_h[tid]       = h_cur[tid];
            lds_h[256 + tid] = h_cur[256 + tid];
            __syncthreads();
        }
    }
}

// ---------------------------------------------------------------------------
// Elementwise helpers
// ---------------------------------------------------------------------------
__global__ void zero_u32(unsigned int* p, int n) {
    int i = blockIdx.x * blockDim.x + threadIdx.x;
    if (i < n) p[i] = 0u;
}
__global__ void bias_sum(const float* a, const float* b, float* o, int n) {
    int i = blockIdx.x * blockDim.x + threadIdx.x;
    if (i < n) o[i] = a[i] + b[i];
}
__global__ void f2bf_flat(const float* __restrict__ s, unsigned short* __restrict__ d, int n) {
    int i = blockIdx.x * blockDim.x + threadIdx.x;
    if (i < n) d[i] = f2bf(s[i]);
}
__global__ void pad2d_f2bf(const float* __restrict__ s, int R, int C,
                           unsigned short* __restrict__ d, int Rp, int Cp) {
    int i = blockIdx.x * blockDim.x + threadIdx.x;
    if (i >= Rp * Cp) return;
    int r = i / Cp, c = i - r * Cp;
    d[i] = (r < R && c < C) ? f2bf(s[(size_t)r * C + c]) : (unsigned short)0;
}
__global__ void padvec_f32(const float* __restrict__ s, int n, float* __restrict__ d, int np) {
    int i = blockIdx.x * blockDim.x + threadIdx.x;
    if (i < np) d[i] = (i < n) ? s[i] : 0.f;
}
__global__ void chars_transpose(const float* __restrict__ chars, unsigned short* __restrict__ d) {
    int i = blockIdx.x * blockDim.x + threadIdx.x;     // over 80*512
    if (i >= NC * U_LEN) return;
    int c = i / U_LEN, u = i - c * U_LEN;
    d[i] = f2bf(chars[(size_t)u * NC + c]);
}
// out[t][j] = b1[j] + b2[j] (+ x[t] @ W[j]) (+ hadd[t][j])   (W is 512x3)
__global__ void fused_affine3(const float* __restrict__ x, const float* __restrict__ W,
                              const float* __restrict__ b1, const float* __restrict__ b2,
                              const float* __restrict__ hadd, float* __restrict__ out, int total) {
    int i = blockIdx.x * blockDim.x + threadIdx.x;
    if (i >= total) return;
    int t = i >> 9, j = i & 511;
    float v = b1[j] + b2[j];
    if (W) {
        const float* xr = x + t * 3;
        const float* wr = W + j * 3;
        v += xr[0] * wr[0] + xr[1] * wr[1] + xr[2] * wr[2];
    }
    if (hadd) v += hadd[i];
    out[i] = v;
}

// ---------------------------------------------------------------------------
// Attention window: per t, phi[t][u] = sum_k alpha_k exp(-beta_k (kappa_k - u)^2)
// (kappa is a cumsum over K within the row -> fully parallel over t).
// ---------------------------------------------------------------------------
__global__ void attention_phi(const float* __restrict__ p,          // T x 32
                              unsigned short* __restrict__ phi) {   // T x 512 bf16
    int t = blockIdx.x;
    int tid = threadIdx.x;               // 64
    const float* pr = p + (size_t)t * 32;
    float a[KMIX], b[KMIX], kap[KMIX];
    float run = 0.f;
#pragma unroll
    for (int k = 0; k < KMIX; ++k) {
        a[k] = __expf(pr[k]);
        b[k] = __expf(pr[KMIX + k]);
        run += __expf(pr[2 * KMIX + k]);
        kap[k] = run;
    }
#pragma unroll
    for (int i = 0; i < 8; ++i) {
        int u = i * 64 + tid;
        float uu = (float)u;
        float s = 0.f;
#pragma unroll
        for (int k = 0; k < KMIX; ++k) {
            float d = kap[k] - uu;
            s += a[k] * __expf(-b[k] * d * d);
        }
        phi[(size_t)t * U_LEN + u] = f2bf(s);
    }
}

// ---------------------------------------------------------------------------
// Loss: softmax of pi_hat over TIME axis (cols 1..20 of o, rows 0..4094),
// then per-row bivariate mixture density + eos terms, deterministic reduction.
// ---------------------------------------------------------------------------
__global__ void pi_colstats(const float* __restrict__ o,
                            float* __restrict__ pimax, float* __restrict__ pisum) {
    __shared__ float red[256];
    int m = blockIdx.x, tid = threadIdx.x;
    float mx = -1e30f;
    for (int t = tid; t < T_STEPS - 1; t += 256) mx = fmaxf(mx, o[(size_t)t * 128 + 1 + m]);
    red[tid] = mx; __syncthreads();
    for (int s = 128; s > 0; s >>= 1) { if (tid < s) red[tid] = fmaxf(red[tid], red[tid + s]); __syncthreads(); }
    float cmax = red[0];
    __syncthreads();
    float sm = 0.f;
    for (int t = tid; t < T_STEPS - 1; t += 256) sm += expf(o[(size_t)t * 128 + 1 + m] - cmax);
    red[tid] = sm; __syncthreads();
    for (int s = 128; s > 0; s >>= 1) { if (tid < s) red[tid] += red[tid + s]; __syncthreads(); }
    if (tid == 0) { pimax[m] = cmax; pisum[m] = red[0]; }
}

__global__ void row_loss(const float* __restrict__ o, const float* __restrict__ hwrt,
                         const float* __restrict__ pimax, const float* __restrict__ pisum,
                         float* __restrict__ partials) {
    __shared__ float rl[128], re[128];
    int t = blockIdx.x * 128 + threadIdx.x;
    float lp = 0.f, le = 0.f;
    if (t < T_STEPS - 1) {
        const float* orow = o + (size_t)t * 128;
        float teos = hwrt[(t + 1) * 3 + 0];
        float x1   = hwrt[(t + 1) * 3 + 1];
        float x2   = hwrt[(t + 1) * 3 + 2];
        float s = 0.f;
        const float TWO_PI = 6.283185307179586f;
        for (int m = 0; m < MMIX; ++m) {
            float pi  = expf(orow[1 + m] - pimax[m]) / pisum[m];
            float mu1 = orow[1 + 1 * MMIX + m];
            float mu2 = orow[1 + 2 * MMIX + m];
            float s1  = expf(orow[1 + 3 * MMIX + m]);
            float s2  = expf(orow[1 + 4 * MMIX + m]);
            float rho = tanhf(orow[1 + 5 * MMIX + m]);
            float z1 = (x1 - mu1) / s1, z2 = (x2 - mu2) / s2;
            float Z  = z1 * z1 + z2 * z2 - 2.f * rho * z1 * z2;
            float om = 1.f - rho * rho;
            float Nv = expf(-Z / (2.f * om)) / (TWO_PI * s1 * s2 * sqrtf(om));
            s += pi * Nv;
        }
        lp = -logf(s + EPS_F);
        float e = 1.f / (1.f + expf(-orow[0]));
        le = -(teos * logf(e + EPS_F) + (1.f - teos) * logf(1.f - e + EPS_F));
    }
    rl[threadIdx.x] = lp; re[threadIdx.x] = le;
    __syncthreads();
    for (int s = 64; s > 0; s >>= 1) {
        if (threadIdx.x < s) { rl[threadIdx.x] += rl[threadIdx.x + s]; re[threadIdx.x] += re[threadIdx.x + s]; }
        __syncthreads();
    }
    if (threadIdx.x == 0) { partials[blockIdx.x * 2] = rl[0]; partials[blockIdx.x * 2 + 1] = re[0]; }
}

__global__ void final_sum(const float* __restrict__ partials, int nb, float* __restrict__ out) {
    if (blockIdx.x == 0 && threadIdx.x == 0) {
        float lp = 0.f, le = 0.f;
        for (int i = 0; i < nb; ++i) { lp += partials[2 * i]; le += partials[2 * i + 1]; }
        out[0] = lp; out[1] = le; out[2] = lp + le;
    }
}

// ---------------------------------------------------------------------------
// Host side
// ---------------------------------------------------------------------------
extern "C" void kernel_launch(void* const* d_in, const int* in_sizes, int n_in,
                              void* d_out, int out_size, void* d_ws, size_t ws_size,
                              hipStream_t stream) {
    (void)in_sizes; (void)n_in; (void)out_size; (void)ws_size;
    const float* hwrt   = (const float*)d_in[0];
    const float* chars  = (const float*)d_in[1];
    const float* in_w   = (const float*)d_in[2];
    const float* in_b   = (const float*)d_in[3];
    const float* sk2_w  = (const float*)d_in[4];
    const float* sk2_b  = (const float*)d_in[5];
    const float* sk3_w  = (const float*)d_in[6];
    const float* sk3_b  = (const float*)d_in[7];
    const float* wt1_b  = (const float*)d_in[9];     // w_prev == 0 -> only bias survives
    const float* wt2_w  = (const float*)d_in[10];
    const float* wt2_b  = (const float*)d_in[11];
    const float* wt3_w  = (const float*)d_in[12];
    const float* wt3_b  = (const float*)d_in[13];
    const float* so1_w  = (const float*)d_in[14];
    const float* so1_b  = (const float*)d_in[15];
    const float* so2_w  = (const float*)d_in[16];
    const float* so2_b  = (const float*)d_in[17];
    const float* att_w  = (const float*)d_in[18];
    const float* att_b  = (const float*)d_in[19];
    const float* out_w  = (const float*)d_in[20];
    const float* out_b  = (const float*)d_in[21];
    const float* wih[3] = { (const float*)d_in[22], (const float*)d_in[26], (const float*)d_in[30] };
    const float* whh[3] = { (const float*)d_in[23], (const float*)d_in[27], (const float*)d_in[31] };
    const float* bih[3] = { (const float*)d_in[24], (const float*)d_in[28], (const float*)d_in[32] };
    const float* bhh[3] = { (const float*)d_in[25], (const float*)d_in[29], (const float*)d_in[33] };

    char* ws = (char*)d_ws;
    size_t off = 0;
    auto alloc = [&](size_t bytes) -> void* {
        void* p = ws + off;
        off = (off + bytes + 255) & ~(size_t)255;
        return p;
    };
    float* xg    = (float*)alloc((size_t)T_STEPS * GATES * 4);
    float* h1    = (float*)alloc((size_t)T_STEPS * HID * 4);
    float* h2    = (float*)alloc((size_t)T_STEPS * HID * 4);
    float* h3    = (float*)alloc((size_t)T_STEPS * HID * 4);
    float* act   = (float*)alloc((size_t)T_STEPS * HID * 4);
    float* wbuf  = (float*)alloc((size_t)T_STEPS * NC * 4);
    float* p32   = (float*)alloc((size_t)T_STEPS * 32 * 4);
    float* o32   = (float*)alloc((size_t)T_STEPS * 128 * 4);
    unsigned short* abf    = (unsigned short*)alloc((size_t)T_STEPS * HID * 2);
    unsigned short* h1bf   = (unsigned short*)alloc((size_t)T_STEPS * HID * 2);
    unsigned short* h2bf   = (unsigned short*)alloc((size_t)T_STEPS * HID * 2);
    unsigned short* phibf  = (unsigned short*)alloc((size_t)T_STEPS * U_LEN * 2);
    unsigned short* w96bf  = (unsigned short*)alloc((size_t)T_STEPS * 96 * 2);
    unsigned short* wihbf  = (unsigned short*)alloc((size_t)GATES * HID * 2);
    unsigned short* sobf   = (unsigned short*)alloc((size_t)HID * HID * 2);
    unsigned short* attwbf = (unsigned short*)alloc((size_t)32 * HID * 2);
    unsigned short* outwbf = (unsigned short*)alloc((size_t)128 * HID * 2);
    unsigned short* wt2bf  = (unsigned short*)alloc((size_t)HID * 96 * 2);
    unsigned short* wt3bf  = (unsigned short*)alloc((size_t)HID * 96 * 2);
    unsigned short* charsT = (unsigned short*)alloc((size_t)NC * U_LEN * 2);
    float* bsum1  = (float*)alloc(GATES * 4);
    float* bsum2  = (float*)alloc(GATES * 4);
    float* bsum3  = (float*)alloc(GATES * 4);
    float* attb32 = (float*)alloc(32 * 4);
    float* outb128= (float*)alloc(128 * 4);
    float* hcur   = (float*)alloc(HID * 4);
    unsigned int* syncs = (unsigned int*)alloc(8 * 4);
    float* pimax  = (float*)alloc(MMIX * 4);
    float* pisum  = (float*)alloc(MMIX * 4);
    float* partials = (float*)alloc(64 * 4);

    const float* bsum[3] = { bsum1, bsum2, bsum3 };

    auto conv = [&](const float* s, unsigned short* d, int n) {
        f2bf_flat<<<(n + 255) / 256, 256, 0, stream>>>(s, d, n);
    };
    // NT-blocked dispatch: Mtiles is always 256 (T/16).
    auto gemm = [&](const unsigned short* A, int lda, const unsigned short* B, int ldb,
                    const float* bias, const float* Cadd, int ldc, float* D, int ldd,
                    int Ntiles, int K) {
        if ((Ntiles & 3) == 0) {
            int g = Ntiles / 4;
            wmma_gemm<4><<<(256 * g) / 4, 128, 0, stream>>>(A, lda, B, ldb, bias, Cadd, ldc,
                                                            D, ldd, g, K);
        } else if ((Ntiles & 1) == 0) {
            int g = Ntiles / 2;
            wmma_gemm<2><<<(256 * g) / 4, 128, 0, stream>>>(A, lda, B, ldb, bias, Cadd, ldc,
                                                            D, ldd, g, K);
        } else {
            wmma_gemm<1><<<(256 * Ntiles) / 4, 128, 0, stream>>>(A, lda, B, ldb, bias, Cadd, ldc,
                                                                 D, ldd, Ntiles, K);
        }
    };
    const size_t LSTM_LDS = 256u * 256u * 4u + 512u * 4u + 256u * 4u;   // 265 KB < 320 KB/WGP

    // ---- one-time prep ----
    zero_u32<<<1, 32, 0, stream>>>(syncs, 8);
    for (int l = 0; l < 3; ++l)
        bias_sum<<<8, 256, 0, stream>>>(bih[l], bhh[l], (float*)bsum[l], GATES);
    pad2d_f2bf<<<(32 * HID + 255) / 256, 256, 0, stream>>>(att_w, 30, HID, attwbf, 32, HID);
    pad2d_f2bf<<<(128 * HID + 255) / 256, 256, 0, stream>>>(out_w, 121, HID, outwbf, 128, HID);
    pad2d_f2bf<<<(HID * 96 + 255) / 256, 256, 0, stream>>>(wt2_w, HID, NC, wt2bf, HID, 96);
    pad2d_f2bf<<<(HID * 96 + 255) / 256, 256, 0, stream>>>(wt3_w, HID, NC, wt3bf, HID, 96);
    padvec_f32<<<1, 32, 0, stream>>>(att_b, 30, attb32, 32);
    padvec_f32<<<1, 128, 0, stream>>>(out_b, 121, outb128, 128);
    chars_transpose<<<(NC * U_LEN + 255) / 256, 256, 0, stream>>>(chars, charsT);

    // ---- LSTM 1 ----
    fused_affine3<<<(T_STEPS * HID) / 256, 256, 0, stream>>>(hwrt, in_w, in_b, wt1_b, nullptr,
                                                             act, T_STEPS * HID);
    conv(act, abf, T_STEPS * HID);
    conv(wih[0], wihbf, GATES * HID);
    gemm(abf, HID, wihbf, HID, bsum[0], nullptr, 0, xg, GATES, GATES / 16, HID);
    lstm_scan<<<8, 256, LSTM_LDS, stream>>>(xg, whh[0], h1, hcur, syncs + 0);
    conv(h1, h1bf, T_STEPS * HID);

    // ---- attention ----
    gemm(h1bf, HID, attwbf, HID, attb32, nullptr, 0, p32, 32, 2, HID);
    attention_phi<<<T_STEPS, 64, 0, stream>>>(p32, phibf);
    gemm(phibf, U_LEN, charsT, U_LEN, nullptr, nullptr, 0, wbuf, NC, NC / 16, U_LEN);
    pad2d_f2bf<<<(T_STEPS * 96 + 255) / 256, 256, 0, stream>>>(wbuf, T_STEPS, NC, w96bf, T_STEPS, 96);

    // ---- LSTM 2 ----
    fused_affine3<<<(T_STEPS * HID) / 256, 256, 0, stream>>>(hwrt, sk2_w, sk2_b, wt2_b, h1,
                                                             act, T_STEPS * HID);
    gemm(w96bf, 96, wt2bf, 96, nullptr, act, HID, act, HID, HID / 16, 96);
    conv(act, abf, T_STEPS * HID);
    conv(wih[1], wihbf, GATES * HID);
    gemm(abf, HID, wihbf, HID, bsum[1], nullptr, 0, xg, GATES, GATES / 16, HID);
    lstm_scan<<<8, 256, LSTM_LDS, stream>>>(xg, whh[1], h2, hcur, syncs + 1);
    conv(h2, h2bf, T_STEPS * HID);

    // ---- LSTM 3 ----
    fused_affine3<<<(T_STEPS * HID) / 256, 256, 0, stream>>>(hwrt, sk3_w, sk3_b, wt3_b, h2,
                                                             act, T_STEPS * HID);
    gemm(w96bf, 96, wt3bf, 96, nullptr, act, HID, act, HID, HID / 16, 96);
    conv(act, abf, T_STEPS * HID);
    conv(wih[2], wihbf, GATES * HID);
    gemm(abf, HID, wihbf, HID, bsum[2], nullptr, 0, xg, GATES, GATES / 16, HID);
    lstm_scan<<<8, 256, LSTM_LDS, stream>>>(xg, whh[2], h3, hcur, syncs + 2);

    // ---- output head: ho = h3 + so2(h2) + so1(h1) ----
    fused_affine3<<<(T_STEPS * HID) / 256, 256, 0, stream>>>(nullptr, nullptr, so1_b, so2_b, h3,
                                                             act, T_STEPS * HID);
    conv(so2_w, sobf, HID * HID);
    gemm(h2bf, HID, sobf, HID, nullptr, act, HID, act, HID, HID / 16, HID);
    conv(so1_w, sobf, HID * HID);
    gemm(h1bf, HID, sobf, HID, nullptr, act, HID, act, HID, HID / 16, HID);
    conv(act, abf, T_STEPS * HID);
    gemm(abf, HID, outwbf, HID, outb128, nullptr, 0, o32, 128, 8, HID);

    // ---- loss ----
    pi_colstats<<<MMIX, 256, 0, stream>>>(o32, pimax, pisum);
    row_loss<<<32, 128, 0, stream>>>(o32, hwrt, pimax, pisum, partials);
    final_sum<<<1, 1, 0, stream>>>(partials, 32, (float*)d_out);
}
